// DeformConvBlock_9637906612435
// MI455X (gfx1250) — matrix operand order repaired and checked by reading, not compiled, not run on hardware
//
#include <hip/hip_runtime.h>

typedef __attribute__((ext_vector_type(16))) _Float16 v16h;
typedef __attribute__((ext_vector_type(8)))  float    v8f;

#define B_    8
#define CIN   64
#define COUT  64
#define H_    128
#define W_    128
#define HW    (H_*W_)
#define K2_   9
#define CK    576      // CIN * K2  (= 18 k-tiles of 32)
#define KT    18
#define TN    32       // pixels per workgroup in the fused sample+GEMM kernel
#define SROW  592      // LDS row stride in halves (576 + 16 pad; 1184B keeps 32B align & breaks conflicts)

// ---------------------------------------------------------------------------
// Kernel 0: pack w_def (Cout=64 x CK=576, f32) into f16 A-matrix fragments and
// zero the BN accumulators. A-fragment layout (16-bit A 16x32, contiguous k per lane half):
//   lane = half*16 + m  (half = K sub-range selector, m = row within 16)
//   vgpr v holds K pair: kk = half*16 + v*2 (contiguous k)
// Stored as wpack[((kt*4 + mt)*32 + lane)*8 + v] (u32 = packed f16 pair) so the
// GEMM loads one contiguous 32-byte v16h per lane per k-tile.
// ---------------------------------------------------------------------------
__global__ void pack_wdef_kernel(const float* __restrict__ wdef,
                                 unsigned int* __restrict__ wpack,
                                 float* __restrict__ gsum,
                                 float* __restrict__ gsumsq) {
    int idx = blockIdx.x * 256 + threadIdx.x;   // exactly 18*4*32*8 = 18432 threads
    if (idx < 64)       gsum[idx]       = 0.0f;
    else if (idx < 128) gsumsq[idx-64]  = 0.0f;

    int v    = idx & 7;
    int lane = (idx >> 3)  & 31;
    int mt   = (idx >> 8)  & 3;
    int kt   =  idx >> 10;
    int hh   = lane >> 4;
    int m    = lane & 15;
    int o    = mt*16 + m;
    int kk   = kt*32 + hh*16 + v*2;
    union { _Float16 h[2]; unsigned int u; } cv;
    cv.h[0] = (_Float16)wdef[o*CK + kk];
    cv.h[1] = (_Float16)wdef[o*CK + kk + 1];
    wpack[idx] = cv.u;
}

// ---------------------------------------------------------------------------
// Kernel 1: offset conv  (3x3, Cin=64 -> 18 ch), LDS-tiled 16x16 spatial tile
// with 18x18 halo, one input channel at a time. off layout: [b][j][y][x].
// ---------------------------------------------------------------------------
__global__ void __launch_bounds__(256) offset_conv_kernel(
    const float* __restrict__ x, const float* __restrict__ woff,
    const float* __restrict__ boff, float* __restrict__ off) {
    __shared__ float xt[18*18];
    __shared__ float wl[18*9];
    const int b   = blockIdx.z;
    const int ty0 = blockIdx.y * 16;
    const int tx0 = blockIdx.x * 16;
    const int tid = threadIdx.x;
    const int ty  = tid >> 4, tx = tid & 15;

    float acc[18];
#pragma unroll
    for (int j = 0; j < 18; ++j) acc[j] = boff[j];

    for (int c = 0; c < CIN; ++c) {
        for (int i = tid; i < 18*18; i += 256) {
            int ly = i / 18, lx = i % 18;
            int gy = ty0 + ly - 1, gx = tx0 + lx - 1;
            float v = 0.0f;
            if (gy >= 0 && gy < H_ && gx >= 0 && gx < W_)
                v = x[((b*CIN + c)*H_ + gy)*W_ + gx];
            xt[i] = v;
        }
        if (tid < 162)  // woff layout (18, 64, 3, 3)
            wl[tid] = woff[(tid/9)*CIN*9 + c*9 + (tid%9)];
        __syncthreads();
#pragma unroll
        for (int j = 0; j < 18; ++j) {
            float s = 0.0f;
#pragma unroll
            for (int ky = 0; ky < 3; ++ky)
#pragma unroll
                for (int kx = 0; kx < 3; ++kx)
                    s += xt[(ty+ky)*18 + (tx+kx)] * wl[j*9 + ky*3 + kx];
            acc[j] += s;
        }
        __syncthreads();
    }
    const int y = ty0 + ty, xq = tx0 + tx;
#pragma unroll
    for (int j = 0; j < 18; ++j)
        off[((b*18 + j)*H_ + y)*W_ + xq] = acc[j];
}

// ---------------------------------------------------------------------------
// Kernel 2: fused bilinear sampling + WMMA GEMM + BN partial reduction.
// Workgroup = 256 threads = 8 waves; tile = 32 consecutive pixels (same b, y;
// 32 | W so a tile never crosses a row) x all 64 output channels.
//
// Phase A: gather-sample S[p][ck] (ck = c*9+k) into LDS as f16.
//          64 threads per (p,k) pair (one per input channel c); 72 passes.
// Phase B: wave w -> (mt = w&3 Cout-tile, nt = w>>2 pixel-tile).
//          18x v_wmma_f32_16x16x32_f16, A from wpack (global, L2-hot),
//          B = contiguous 32B LDS read per lane:
//            lane = half*16 + n, read &S[(nt*16+n)*SROW + kt*32 + half*16].
// ---------------------------------------------------------------------------
__global__ void __launch_bounds__(256) deform_gemm_kernel(
    const float* __restrict__ x, const float* __restrict__ off,
    const unsigned int* __restrict__ wpack, const float* __restrict__ bdef,
    float* __restrict__ out, float* __restrict__ gsum,
    float* __restrict__ gsumsq) {
    __shared__ __align__(32) _Float16 S[TN * SROW];
    __shared__ float lsum[COUT];
    __shared__ float lsq[COUT];

    const int tid = threadIdx.x;
    if (tid < COUT) { lsum[tid] = 0.0f; lsq[tid] = 0.0f; }

    const int n0  = blockIdx.x * TN;        // global pixel base (b*HW + y*W + x0)
    const int b   = n0 / HW;
    const int rem = n0 % HW;
    const int y   = rem / W_;
    const int x0  = rem % W_;
    const float* __restrict__ xb = x + (size_t)b * CIN * HW;

    // ---------------- Phase A: bilinear sampling into LDS ----------------
    const int c    = tid & 63;
    const int quad = tid >> 6;
    for (int pass = 0; pass < 72; ++pass) {
        int pr = pass*4 + quad;             // 0..287 = 32 pixels x 9 taps
        int p  = pr / 9,  k = pr % 9;
        int px = x0 + p;
        float dy = off[((b*18 + 2*k    )*H_ + y)*W_ + px];
        float dx = off[((b*18 + 2*k + 1)*H_ + y)*W_ + px];
        float py  = (float)y  - 1.0f + (float)(k/3) + dy;
        float pxx = (float)px - 1.0f + (float)(k%3) + dx;
        float y0f = floorf(py),  x0f = floorf(pxx);
        float wy1 = py - y0f,    wx1 = pxx - x0f;
        float wy0 = 1.0f - wy1,  wx0 = 1.0f - wx1;
        int   y0i = (int)y0f,    x0i = (int)x0f;
        bool vy0 = (y0f >= 0.0f)        && (y0f <= (float)(H_-1));
        bool vy1 = (y0f + 1.0f >= 0.0f) && (y0f + 1.0f <= (float)(H_-1));
        bool vx0 = (x0f >= 0.0f)        && (x0f <= (float)(W_-1));
        bool vx1 = (x0f + 1.0f >= 0.0f) && (x0f + 1.0f <= (float)(W_-1));
        int yc0 = min(max(y0i,     0), H_-1), yc1 = min(max(y0i + 1, 0), H_-1);
        int xc0 = min(max(x0i,     0), W_-1), xc1 = min(max(x0i + 1, 0), W_-1);
        const float* __restrict__ xc = xb + (size_t)c * HW;
        float v00 = (vy0 && vx0) ? xc[yc0*W_ + xc0] : 0.0f;
        float v01 = (vy0 && vx1) ? xc[yc0*W_ + xc1] : 0.0f;
        float v10 = (vy1 && vx0) ? xc[yc1*W_ + xc0] : 0.0f;
        float v11 = (vy1 && vx1) ? xc[yc1*W_ + xc1] : 0.0f;
        float val = wy0*wx0*v00 + wy0*wx1*v01 + wy1*wx0*v10 + wy1*wx1*v11;
        S[p*SROW + c*9 + k] = (_Float16)val;
    }
    __syncthreads();

    // ---------------- Phase B: 18-step WMMA GEMM ----------------
    const int wave = tid >> 5, lane = tid & 31;
    const int mt = wave & 3, nt = wave >> 2;
    const int hh = lane >> 4, ln = lane & 15;

    v8f acc = {};
    const v16h* __restrict__ A = (const v16h*)wpack;
#pragma unroll
    for (int kt = 0; kt < KT; ++kt) {
        v16h afrag = A[(kt*4 + mt)*32 + lane];
        v16h bfrag = *(const v16h*)&S[(nt*16 + ln)*SROW + kt*32 + hh*16];
        acc = __builtin_amdgcn_wmma_f32_16x16x32_f16(
                  false, afrag, false, bfrag, (short)0, acc, false, false);
    }

    // store pre-BN output + BN partials. C/D layout: M = vgpr + 8*hh, N = ln.
    const int n  = nt*16 + ln;
    const int sp = (n0 + n) % HW;
#pragma unroll
    for (int r = 0; r < 8; ++r) {
        int o = mt*16 + hh*8 + r;
        float v = acc[r] + bdef[o];
        out[((size_t)b*COUT + o)*HW + sp] = v;
        atomicAdd(&lsum[o], v);
        atomicAdd(&lsq[o],  v*v);
    }
    __syncthreads();
    if (tid < COUT) {
        atomicAdd(&gsum[tid],   lsum[tid]);
        atomicAdd(&gsumsq[tid], lsq[tid]);
    }
}

// ---------------------------------------------------------------------------
// Kernel 3: BN finalize + ReLU, in place on d_out.
// ---------------------------------------------------------------------------
__global__ void bn_relu_kernel(float* __restrict__ out,
                               const float* __restrict__ gsum,
                               const float* __restrict__ gsumsq,
                               const float* __restrict__ gamma,
                               const float* __restrict__ beta) {
    long long idx = (long long)blockIdx.x * blockDim.x + threadIdx.x;
    const long long total = (long long)B_ * COUT * HW;
    if (idx >= total) return;
    int o = (int)((idx / HW) % COUT);
    const float N = (float)(B_ * HW);
    float mean = gsum[o] / N;
    float var  = gsumsq[o] / N - mean*mean;
    float rs   = rsqrtf(var + 1e-5f);
    float v    = (out[idx] - mean) * rs * gamma[o] + beta[o];
    out[idx]   = v > 0.0f ? v : 0.0f;
}

// ---------------------------------------------------------------------------
// Launch: ws layout = [off: 9,437,184 B][wpack: 73,728 B][gsum:256B][gsumsq:256B]
// ---------------------------------------------------------------------------
extern "C" void kernel_launch(void* const* d_in, const int* in_sizes, int n_in,
                              void* d_out, int out_size, void* d_ws, size_t ws_size,
                              hipStream_t stream) {
    const float* x     = (const float*)d_in[0];
    const float* woff  = (const float*)d_in[1];
    const float* boff  = (const float*)d_in[2];
    const float* wdef  = (const float*)d_in[3];
    const float* bdef  = (const float*)d_in[4];
    const float* gamma = (const float*)d_in[5];
    const float* beta  = (const float*)d_in[6];
    float* out = (float*)d_out;

    char* ws = (char*)d_ws;
    float*        off    = (float*)ws;                               // 8*18*128*128 f32
    unsigned int* wpack  = (unsigned int*)(ws + 9437184);            // 18432 u32
    float*        gsum   = (float*)(ws + 9437184 + 73728);           // 64 f32
    float*        gsumsq = gsum + 64;                                // 64 f32

    pack_wdef_kernel<<<72, 256, 0, stream>>>(wdef, wpack, gsum, gsumsq);

    dim3 g1(W_/16, H_/16, B_);
    offset_conv_kernel<<<g1, 256, 0, stream>>>(x, woff, boff, off);

    deform_gemm_kernel<<<(B_*HW)/TN, 256, 0, stream>>>(x, off, wpack, bdef,
                                                       out, gsum, gsumsq);

    int total = B_*COUT*HW;
    bn_relu_kernel<<<(total + 255)/256, 256, 0, stream>>>(out, gsum, gsumsq,
                                                          gamma, beta);
}
